// ROUNDModel_13469017440914
// MI455X (gfx1250) — compile-verified, matrix operand order checked by reading, not run on hardware
//
#include <hip/hip_runtime.h>
#include <hip/hip_bf16.h>

typedef __attribute__((ext_vector_type(16))) _Float16 v16h;
typedef __attribute__((ext_vector_type(8)))  _Float16 v8h;
typedef __attribute__((ext_vector_type(8)))  float    v8f;

#define B_ROWS  131072
#define HDIM    128
#define STEPS   12
#define WSTRIDE 264   // halves per LDS row: 256 data + 8 pad = 528 B (16B aligned, 4-bank rotate)

// Load one 16x32-f16 WMMA operand fragment (A or B) from an LDS row laid out
// as 256 interleaved halves with stride WSTRIDE.
//   lane role:  n16 = lane&15 selects the row (M for A, N for B),
//               hi  = lane>>4 selects which K-halves this lane holds.
// A-layout (ISA 7.12.2): lane<16 holds K={0..7,16..23}, lane>=16 K={8..15,24..31};
// B assumed symmetric (lane indexes N, holds K values).
__device__ __forceinline__ v16h ld_frag(const _Float16* rowbase, int kb, int hi) {
    const _Float16* p = rowbase + kb * 32 + hi * 8;
    v8h lo = *(const v8h*)(p);
    v8h hh = *(const v8h*)(p + 16);
    return __builtin_shufflevector(lo, hh, 0,1,2,3,4,5,6,7,8,9,10,11,12,13,14,15);
}

// acc[nt] (+)= feats[16 x 256] @ W[nt-th 16-col slab].T   via v_wmma_f32_16x16x32_f16
__device__ __forceinline__ void gemm_16x256(const _Float16* feats, const _Float16* W,
                                            int n16, int hi, v8f acc[8]) {
    v16h a[8];
    const _Float16* arow = feats + n16 * WSTRIDE;
#pragma unroll
    for (int kb = 0; kb < 8; ++kb) a[kb] = ld_frag(arow, kb, hi);
#pragma unroll
    for (int nt = 0; nt < 8; ++nt) {
        const _Float16* brow = W + (nt * 16 + n16) * WSTRIDE;
        v8f c = acc[nt];
#pragma unroll
        for (int kb = 0; kb < 8; ++kb) {
            v16h b = ld_frag(brow, kb, hi);
            c = __builtin_amdgcn_wmma_f32_16x16x32_f16(
                    /*neg_a=*/false, a[kb], /*neg_b=*/false, b,
                    /*c_mod=*/(short)0, c, /*reuse_a=*/false, /*reuse_b=*/false);
        }
        acc[nt] = c;
    }
}

__device__ __forceinline__ unsigned pack_cs(float v) {
    _Float16 h2[2];
    h2[0] = (_Float16)__cosf(v);
    h2[1] = (_Float16)__sinf(v);
    unsigned u;
    __builtin_memcpy(&u, h2, 4);
    return u;
}

__global__ __launch_bounds__(256, 1)
void phase_acc_kernel(const float* __restrict__ x,
                      const float* __restrict__ e_w,
                      const float* __restrict__ e_b,
                      const float* __restrict__ d_w,
                      const float* __restrict__ d_b,
                      const float* __restrict__ r_w,
                      const float* __restrict__ r_b,
                      float* __restrict__ out,
                      float* __restrict__ hstack) {
    extern __shared__ _Float16 lds[];
    _Float16* Wstep  = lds;                          // [128][WSTRIDE] interleaved {cos,sin} weights
    _Float16* Wconst = lds + 128 * WSTRIDE;          // [128][WSTRIDE] interleaved {xp_c,xp_s} weights

    const int tid  = threadIdx.x;
    const int lane = tid & 31;
    const int wave = tid >> 5;
    _Float16* feats = lds + 2 * 128 * WSTRIDE + wave * 16 * WSTRIDE;  // wave-private 16x256 f16

    // ---- one-time: stage interleaved f16 weights into LDS (coalesced f32 reads) ----
    for (int idx = tid; idx < 128 * 128; idx += 256) {
        int n = idx >> 7, j = idx & 127;
        const float* row = d_w + (size_t)n * 512;
        Wstep [n * WSTRIDE + 2 * j]     = (_Float16)row[j];            // cos-part weight
        Wstep [n * WSTRIDE + 2 * j + 1] = (_Float16)row[128 + j];      // sin-part weight
        Wconst[n * WSTRIDE + 2 * j]     = (_Float16)row[256 + j];      // xp_c weight
        Wconst[n * WSTRIDE + 2 * j + 1] = (_Float16)row[384 + j];      // xp_s weight
    }
    __syncthreads();

    const int n16  = lane & 15;
    const int hi   = lane >> 4;
    const int row0 = blockIdx.x * 128 + wave * 16;   // this wave's 16 batch rows

    // ---- embed: p = x*e_w + e_b ; pack {cos p, sin p} into feats ----
    float xv[8];
#pragma unroll
    for (int r = 0; r < 8; ++r) xv[r] = x[row0 + r + 8 * hi];
#pragma unroll
    for (int nt = 0; nt < 8; ++nt) {
        const int n = nt * 16 + n16;
        const float ew = e_w[n], eb = e_b[n];
        unsigned* dst = (unsigned*)(feats + 2 * n);
#pragma unroll
        for (int r = 0; r < 8; ++r) {
            float p = fmaf(xv[r], ew, eb);
            dst[(r + 8 * hi) * (WSTRIDE / 2)] = pack_cs(p);
        }
    }

    // ---- step-invariant constant: cc = [xp_c,xp_s] @ Wconst.T + d_b ----
    v8f cc[8];
#pragma unroll
    for (int nt = 0; nt < 8; ++nt) {
        const float dbv = d_b[nt * 16 + n16];
        v8f c;
#pragma unroll
        for (int i = 0; i < 8; ++i) c[i] = dbv;
        cc[nt] = c;
    }
    gemm_16x256(feats, Wconst, n16, hi, cc);

    // ---- 12-step recurrence, ph register-resident in C/D layout ----
    v8f ph[8];
#pragma unroll
    for (int nt = 0; nt < 8; ++nt)
#pragma unroll
        for (int i = 0; i < 8; ++i) ph[nt][i] = 0.0f;

    for (int s = 0; s < STEPS; ++s) {
        // pack {cos ph, sin ph} (f16, K-interleaved) into wave-private LDS tile
#pragma unroll
        for (int nt = 0; nt < 8; ++nt) {
            unsigned* dst = (unsigned*)(feats + 2 * (nt * 16 + n16));
#pragma unroll
            for (int r = 0; r < 8; ++r)
                dst[(r + 8 * hi) * (WSTRIDE / 2)] = pack_cs(ph[nt][r]);
        }

        // ph_new = (ph + cc) + feats @ Wstep.T   (C initialized with ph+cc)
        v8f acc[8];
#pragma unroll
        for (int nt = 0; nt < 8; ++nt) acc[nt] = ph[nt] + cc[nt];
        gemm_16x256(feats, Wstep, n16, hi, acc);

        // emit Hstack[s, row0:row0+16, :] (coalesced 64B row segments) and carry ph
        float* hrow = hstack + ((size_t)s * B_ROWS + row0) * (size_t)HDIM;
#pragma unroll
        for (int nt = 0; nt < 8; ++nt) {
#pragma unroll
            for (int r = 0; r < 8; ++r)
                hrow[(r + 8 * hi) * HDIM + nt * 16 + n16] = acc[nt][r];
            ph[nt] = acc[nt];
        }
    }

    // ---- readout: out = [cos ph, sin ph, ph] @ r_w.T + r_b ----
    float partial[8];
#pragma unroll
    for (int r = 0; r < 8; ++r) partial[r] = 0.0f;
#pragma unroll
    for (int nt = 0; nt < 8; ++nt) {
        const int n = nt * 16 + n16;
        const float rc = r_w[n], rs = r_w[128 + n], rp = r_w[256 + n];
#pragma unroll
        for (int r = 0; r < 8; ++r) {
            const float v = ph[nt][r];
            partial[r] += __cosf(v) * rc + __sinf(v) * rs + v * rp;
        }
    }
    // reduce the 16 column-lanes per row through wave-private LDS (reuse feats)
    float* scr = (float*)feats;   // 16 x 17 floats < tile size
#pragma unroll
    for (int r = 0; r < 8; ++r) scr[(r + 8 * hi) * 17 + n16] = partial[r];
    if (lane < 16) {
        float ssum = r_b[0];
#pragma unroll
        for (int c = 0; c < 16; ++c) ssum += scr[lane * 17 + c];
        out[row0 + lane] = ssum;
    }
}

extern "C" void kernel_launch(void* const* d_in, const int* in_sizes, int n_in,
                              void* d_out, int out_size, void* d_ws, size_t ws_size,
                              hipStream_t stream) {
    const float* x   = (const float*)d_in[0];
    const float* e_w = (const float*)d_in[1];
    const float* e_b = (const float*)d_in[2];
    const float* d_w = (const float*)d_in[3];
    const float* d_b = (const float*)d_in[4];
    const float* r_w = (const float*)d_in[5];
    const float* r_b = (const float*)d_in[6];
    // d_in[7] = steps (constant 12 in the reference; hardcoded for full unroll)

    float* out    = (float*)d_out;            // [B]
    float* hstack = out + B_ROWS;             // [STEPS, B, H]

    const size_t shmem = (size_t)(2 * 128 * WSTRIDE + 8 * 16 * WSTRIDE) * 2;  // bytes
    dim3 grid(B_ROWS / 128), block(256);
    hipLaunchKernelGGL(phase_acc_kernel, grid, block, shmem, stream,
                       x, e_w, e_b, d_w, d_b, r_w, r_b, out, hstack);
}
